// RMCRNNCell_73727408603625
// MI455X (gfx1250) — compile-verified, matrix operand order checked by reading, not run on hardware
//
#include <hip/hip_runtime.h>
#include <hip/hip_bf16.h>
#include <math.h>

typedef __attribute__((ext_vector_type(16))) __bf16 v16bf;
typedef __attribute__((ext_vector_type(8)))  __bf16 v8bf;
typedef __attribute__((ext_vector_type(8)))  float  v8f;
typedef unsigned int u32x4 __attribute__((ext_vector_type(4)));
typedef int          i32x4 __attribute__((ext_vector_type(4)));
typedef int          i32x8 __attribute__((ext_vector_type(8)));

#if defined(__has_builtin)
#if __has_builtin(__builtin_amdgcn_tensor_load_to_lds) && __has_builtin(__builtin_amdgcn_s_wait_tensorcnt)
#define USE_TDM 1
#else
#define USE_TDM 0
#endif
#else
#define USE_TDM 0
#endif

// ---------------- helpers ----------------
static __device__ __forceinline__ __bf16 to_bf16(float f) {
  union { float f; unsigned u; } in; in.f = f;
  unsigned u = in.u;
  u += 0x7FFFu + ((u >> 16) & 1u);                 // round-to-nearest-even
  union { unsigned short s; __bf16 b; } out; out.s = (unsigned short)(u >> 16);
  return out.b;
}

static __device__ __forceinline__ float gelu_tanh(float x) {
  const float x3 = x * x * x;
  return 0.5f * x * (1.0f + tanhf(0.79788456080286535588f * (x + 0.044715f * x3)));
}

#if USE_TDM
// Issue one TDM descriptor: load a 32(dim0) x rows(dim1) tile of bf16 from a
// row-major matrix (row stride = stride_elems) into LDS at lds_off.
// D# layout per CDNA5 ISA ch.8 (group0 128b, group1 256b; groups 2/3 zero => 2D).
static __device__ __forceinline__ void tdm_load_tile(
    unsigned lds_off, const void* gptr, int stride_elems, int rows) {
  const unsigned long long ga = (unsigned long long)(uintptr_t)gptr;
  u32x4 g0;
  g0[0] = 1u;                                        // count=1, user D#
  g0[1] = lds_off;                                   // lds_addr (bytes)
  g0[2] = (unsigned)(ga & 0xFFFFFFFFull);            // global_addr[31:0]
  g0[3] = (unsigned)((ga >> 32) & 0x1FFFFFFull)      // global_addr[56:32]
        | (2u << 30);                                // type = 2 ("image")
  i32x8 g1;
  g1[0] = 0x00010000;                                // data_size=1 (2 bytes); mask=0
  g1[1] = (int)(((unsigned)stride_elems & 0xFFFFu) << 16);          // tensor_dim0[15:0]
  g1[2] = (int)((((unsigned)stride_elems >> 16) & 0xFFFFu)          // tensor_dim0[31:16]
        | ((unsigned)rows << 16));                                  // tensor_dim1[15:0]
  g1[3] = (int)(32u << 16);                          // tile_dim0 = 32 (K chunk)
  g1[4] = rows;                                      // tile_dim1 = rows
  g1[5] = stride_elems;                              // tensor_dim0_stride[31:0]
  g1[6] = 0;
  g1[7] = 0;
  i32x4 gz = {0, 0, 0, 0};
#if __clang_major__ >= 23
  i32x8 gz8 = {0, 0, 0, 0, 0, 0, 0, 0};
  __builtin_amdgcn_tensor_load_to_lds(g0, g1, gz, gz, gz8, 0);
#else
  __builtin_amdgcn_tensor_load_to_lds(g0, g1, gz, gz, 0);
#endif
}
#endif

// ---------------- conversion kernels ----------------
__global__ void cvt_f32_bf16(const float* __restrict__ in, __bf16* __restrict__ out, int n) {
  int i = blockIdx.x * blockDim.x + threadIdx.x;
  const int stride = gridDim.x * blockDim.x;
  for (; i < n; i += stride) out[i] = to_bf16(in[i]);
}

// transpose + cast: in (K x N) f32 row-major -> out (N x K) bf16 row-major
__global__ void cvt_f32_bf16_T(const float* __restrict__ in, __bf16* __restrict__ out,
                               int K, int N) {
  int i = blockIdx.x * blockDim.x + threadIdx.x;
  const int n = K * N;
  const int stride = gridDim.x * blockDim.x;
  for (; i < n; i += stride) {
    const int k = i / N;
    const int c = i - k * N;
    out[(size_t)c * K + k] = to_bf16(in[i]);
  }
}

// split mlp_kernel (1024 x 2048) into transposed bf16 w1t/w2t (each 1024 x 1024)
__global__ void cvt_split_mlp_T(const float* __restrict__ in, __bf16* __restrict__ w1t,
                                __bf16* __restrict__ w2t) {
  const int n = 1024 * 2048;
  int i = blockIdx.x * blockDim.x + threadIdx.x;
  const int stride = gridDim.x * blockDim.x;
  for (; i < n; i += stride) {
    const int r = i >> 11;
    const int c = i & 2047;
    const __bf16 v = to_bf16(in[i]);
    if (c < 1024) w1t[(size_t)c * 1024 + r] = v;
    else          w2t[(size_t)(c - 1024) * 1024 + r] = v;
  }
}

// rel_emb = rel_pos_table(17 x 1024) @ rel_kernel(1024 x 1024); tiny -> scalar path
__global__ __launch_bounds__(256) void rel_emb_kernel(const float* __restrict__ rel_kernel,
                                                      float* __restrict__ rel_emb) {
  __shared__ float tab[1024];
  const float p = (float)((int)blockIdx.x - 8);
  for (int i = threadIdx.x; i < 1024; i += 256) {
    const float expo = (float)(i - (i & 1)) * (1.0f / 1024.0f);
    const float angle = p / powf(10000.0f, expo);
    tab[i] = (i & 1) ? cosf(angle) : sinf(angle);
  }
  __syncthreads();
  for (int j = threadIdx.x; j < 1024; j += 256) {
    float acc = 0.0f;
    for (int kk = 0; kk < 1024; ++kk)
      acc += tab[kk] * rel_kernel[(size_t)kk * 1024 + j];
    rel_emb[(size_t)blockIdx.x * 1024 + j] = acc;
  }
}

// build mem_plus (B,9,1024) f32+bf16 from state + x; also bf16 copy of memory
__global__ void pack_memplus(const float* __restrict__ state, const float* __restrict__ x,
                             float* __restrict__ mp_f, __bf16* __restrict__ mp_bf,
                             __bf16* __restrict__ mem_bf, int n) {
  int i = blockIdx.x * blockDim.x + threadIdx.x;
  const int stride = gridDim.x * blockDim.x;
  for (; i < n; i += stride) {
    const int b = i / 9216;
    const int r = i - b * 9216;
    const int s = r >> 10;
    const int u = r & 1023;
    float v;
    if (s < 8) {
      const size_t mi = (size_t)b * 8192 + ((size_t)s << 10) + u;
      v = state[mi];
      mem_bf[mi] = to_bf16(v);
    } else {
      v = x[(size_t)b * 1024 + u];
    }
    mp_f[i] = v;
    mp_bf[i] = to_bf16(v);
  }
}

// ---- WMMA bf16 GEMM: C(MxN f32) = A(MxK bf16, row-major) * Bt(NxK bf16, row-major)^T ----
// 128x64 block tile, K-step 32, 128 threads (4 wave32's); each wave: 32 rows x 64 cols
// (8 f32 accumulators). Tiles staged by the Tensor Data Mover with double buffering:
// wave 0 issues the descriptor pair for tile t+1 before the block computes tile t,
// then s_wait_tensorcnt 2 (pairs retire in order => tile t has landed).
__global__ __launch_bounds__(128) void gemm_bf16_wmma(
    const __bf16* __restrict__ A, const __bf16* __restrict__ Bt,
    const float* __restrict__ bias, float* __restrict__ C,
    __bf16* __restrict__ Cbf, int Mrows, int N, int K, int act)
{
  __shared__ __bf16 As[2][128][32];   // ping-pong: 128 rows (M) x 32 (K)
  __shared__ __bf16 Bs[2][64][32];    // ping-pong: 64 rows (N) x 32 (K)
  const int tid  = threadIdx.x;
  const int wave = tid >> 5;
  const int lane = tid & 31;
  const int hf   = lane >> 4;
  const int l16  = lane & 15;
  const int bm = blockIdx.y * 128;
  const int bn = blockIdx.x * 64;
  const int NT = K >> 5;              // K / 32 tiles

  v8f acc[2][4];
#pragma unroll
  for (int rt = 0; rt < 2; ++rt)
#pragma unroll
    for (int nt = 0; nt < 4; ++nt)
#pragma unroll
      for (int j = 0; j < 8; ++j) acc[rt][nt][j] = 0.0f;

#if USE_TDM
  if (tid < 32)  // one wave drives the DMA engine; prologue: tile 0 into buffer 0
  {
    tdm_load_tile((unsigned)(uintptr_t)&As[0][0][0], &A [(size_t)bm * K], K, 128);
    tdm_load_tile((unsigned)(uintptr_t)&Bs[0][0][0], &Bt[(size_t)bn * K], K, 64);
  }
#endif

  for (int t = 0; t < NT; ++t) {
#if USE_TDM
    const int cur = t & 1;
    if (tid < 32) {
      if (t + 1 < NT) {       // issue next tile into the other buffer, then wait for cur
        const int nxt = cur ^ 1;
        const int k0n = (t + 1) << 5;
        tdm_load_tile((unsigned)(uintptr_t)&As[nxt][0][0], &A [(size_t)bm * K + k0n], K, 128);
        tdm_load_tile((unsigned)(uintptr_t)&Bs[nxt][0][0], &Bt[(size_t)bn * K + k0n], K, 64);
        __builtin_amdgcn_s_wait_tensorcnt(2);
      } else {
        __builtin_amdgcn_s_wait_tensorcnt(0);
      }
    }
#else
    const int cur = 0;
    const int k0 = t << 5;
    // fallback: vectorized cooperative loads (row stride K, 128-bit chunks)
#pragma unroll
    for (int c = 0; c < 4; ++c) {            // A tile: 128x32 = 512 chunks
      const int li  = tid + c * 128;
      const int row = li >> 2;
      const int cc  = (li & 3) << 3;
      *(v8bf*)&As[0][row][cc] = *(const v8bf*)&A[(size_t)(bm + row) * K + (k0 + cc)];
    }
#pragma unroll
    for (int c = 0; c < 2; ++c) {            // B tile: 64x32 = 256 chunks
      const int li  = tid + c * 128;
      const int row = li >> 2;
      const int cc  = (li & 3) << 3;
      *(v8bf*)&Bs[0][row][cc] = *(const v8bf*)&Bt[(size_t)(bn + row) * K + (k0 + cc)];
    }
#endif
    __syncthreads();

    // A fragments per ISA 16-bit A 16x32 layout (two contiguous 16B runs per lane)
    union Frag { v16bf v; unsigned u[8]; };
    Frag af[2];
#pragma unroll
    for (int rt = 0; rt < 2; ++rt) {
      const int arow = (wave << 5) + (rt << 4) + l16;
#pragma unroll
      for (int i = 0; i < 8; ++i) {
        const int kk = ((i < 4) ? 0 : 16) + (hf << 3) + ((i & 3) << 1);
        af[rt].u[i] = *(const unsigned*)&As[cur][arow][kk];
      }
    }
#pragma unroll
    for (int nt = 0; nt < 4; ++nt) {
      Frag bfr;
      const int bcol = (nt << 4) + l16;
#pragma unroll
      for (int i = 0; i < 8; ++i) {
        const int kk = ((i < 4) ? 0 : 16) + (hf << 3) + ((i & 3) << 1);
        bfr.u[i] = *(const unsigned*)&Bs[cur][bcol][kk];
      }
      acc[0][nt] = __builtin_amdgcn_wmma_f32_16x16x32_bf16(
          false, af[0].v, false, bfr.v, (short)0, acc[0][nt], false, false);
      acc[1][nt] = __builtin_amdgcn_wmma_f32_16x16x32_bf16(
          false, af[1].v, false, bfr.v, (short)0, acc[1][nt], false, false);
    }
    __syncthreads();
  }

  // epilogue: C/D 16x16 f32 layout -> global, optional bias / GELU / bf16 copy
#pragma unroll
  for (int rt = 0; rt < 2; ++rt) {
#pragma unroll
    for (int nt = 0; nt < 4; ++nt) {
#pragma unroll
      for (int vi = 0; vi < 8; ++vi) {
        const int row = bm + (wave << 5) + (rt << 4) + (hf << 3) + vi;
        const int col = bn + (nt << 4) + l16;
        float val = acc[rt][nt][vi];
        if (bias) val += bias[col];
        if (act == 1) val = gelu_tanh(val);
        const size_t o = (size_t)row * N + col;
        if (C)   C[o]   = val;
        if (Cbf) Cbf[o] = to_bf16(val);
      }
    }
  }
  (void)Mrows;
}

// ---------------- attention: one (batch, head) per block ----------------
__global__ __launch_bounds__(128) void attn_kernel(const float* __restrict__ qkv,
                                                   const float* __restrict__ rel_emb,
                                                   float* __restrict__ att_out) {
  __shared__ float qs[9][64], ks[9][64], vs[9][64], sc[9][9];
  const int h = blockIdx.x;
  const int b = blockIdx.y;
  const int tid = threadIdx.x;
  for (int i = tid; i < 576; i += 128) {
    const int s = i >> 6, d = i & 63;
    const size_t base = ((size_t)(b * 9 + s)) * 3072 + h * 64 + d;
    qs[s][d] = qkv[base] * 0.125f;       // 1/sqrt(64)
    ks[s][d] = qkv[base + 1024];
    vs[s][d] = qkv[base + 2048];
  }
  __syncthreads();
  if (tid < 81) {                        // q·k + q·rel == q·(k + rel_row)
    const int qi = tid / 9, ki = tid - qi * 9;
    const float* re = rel_emb + (size_t)(qi - ki + 8) * 1024 + h * 64;
    float acc = 0.0f;
#pragma unroll 8
    for (int d = 0; d < 64; ++d) acc += qs[qi][d] * (ks[ki][d] + re[d]);
    sc[qi][ki] = acc;
  }
  __syncthreads();
  if (tid < 9) {
    float m = sc[tid][0];
#pragma unroll
    for (int k = 1; k < 9; ++k) m = fmaxf(m, sc[tid][k]);
    float sum = 0.0f;
#pragma unroll
    for (int k = 0; k < 9; ++k) { const float e = expf(sc[tid][k] - m); sc[tid][k] = e; sum += e; }
    const float inv = 1.0f / sum;
#pragma unroll
    for (int k = 0; k < 9; ++k) sc[tid][k] *= inv;
  }
  __syncthreads();
  for (int i = tid; i < 576; i += 128) {
    const int qi = i >> 6, d = i & 63;
    float acc = 0.0f;
#pragma unroll
    for (int k = 0; k < 9; ++k) acc += sc[qi][k] * vs[k][d];
    att_out[((size_t)(b * 9 + qi)) * 1024 + h * 64 + d] = acc;
  }
}

// ---------------- LayerNorm(Xa + Xb), one row (1024) per 256-thread block ----------------
__global__ __launch_bounds__(256) void ln_res_kernel(
    const float* __restrict__ Xa, const float* __restrict__ Xb,
    const float* __restrict__ gamma, const float* __restrict__ beta,
    float* __restrict__ Y, __bf16* __restrict__ Ybf, float* __restrict__ out9) {
  __shared__ float red[256];
  const int row = blockIdx.x;
  const int tid = threadIdx.x;
  const size_t base = (size_t)row * 1024;
  float v[4];
  float s = 0.0f;
#pragma unroll
  for (int i = 0; i < 4; ++i) {
    const int c = tid + (i << 8);
    v[i] = Xa[base + c] + Xb[base + c];
    s += v[i];
  }
  red[tid] = s; __syncthreads();
  for (int o = 128; o > 0; o >>= 1) { if (tid < o) red[tid] += red[tid + o]; __syncthreads(); }
  const float mean = red[0] * (1.0f / 1024.0f);
  __syncthreads();
  float ss = 0.0f;
#pragma unroll
  for (int i = 0; i < 4; ++i) { const float d = v[i] - mean; ss += d * d; }
  red[tid] = ss; __syncthreads();
  for (int o = 128; o > 0; o >>= 1) { if (tid < o) red[tid] += red[tid + o]; __syncthreads(); }
  const float inv = rsqrtf(red[0] * (1.0f / 1024.0f) + 1e-5f);
#pragma unroll
  for (int i = 0; i < 4; ++i) {
    const int c = tid + (i << 8);
    const float y = (v[i] - mean) * inv * gamma[c] + beta[c];
    if (Y)   Y[base + c]   = y;
    if (Ybf) Ybf[base + c] = to_bf16(y);
    if (out9 && (row % 9) == 8) out9[(size_t)(row / 9) * 1024 + c] = y;
  }
}

// ---------------- fused gates -> next_state ----------------
__global__ void gate_fuse_kernel(const float* __restrict__ gx, const float* __restrict__ gr,
                                 const float* __restrict__ rbias, const float* __restrict__ state,
                                 const float* __restrict__ h2, float* __restrict__ next_state,
                                 int n) {
  int i = blockIdx.x * blockDim.x + threadIdx.x;
  const int stride = gridDim.x * blockDim.x;
  for (; i < n; i += stride) {
    const int b = i >> 13;
    const int r = i & 8191;
    const int m = r >> 10;
    const int u = r & 1023;
    const size_t gxb = (size_t)b * 2048;
    const size_t grb = (size_t)(b * 8 + m) * 2048;
    const float ig_raw = gx[gxb + u] + gr[grb + u] + rbias[u];
    const float fg_raw = gx[gxb + 1024 + u] + gr[grb + 1024 + u] + rbias[1024 + u] + 1.0f;
    const float ig = fminf(fmaxf(0.2f * ig_raw + 0.5f, 0.0f), 1.0f);
    const float fg = fminf(fmaxf(0.2f * fg_raw + 0.5f, 0.0f), 1.0f);
    const float nm = h2[((size_t)(b * 9 + m)) * 1024 + u];
    next_state[i] = fg * state[i] + ig * tanhf(nm);
  }
}

// ---------------- launch ----------------
extern "C" void kernel_launch(void* const* d_in, const int* in_sizes, int n_in,
                              void* d_out, int out_size, void* d_ws, size_t ws_size,
                              hipStream_t stream) {
  (void)in_sizes; (void)n_in; (void)out_size; (void)ws_size;
  const float* inputs = (const float*)d_in[0];
  const float* state  = (const float*)d_in[1];
  const float* Wi     = (const float*)d_in[2];
  const float* bi     = (const float*)d_in[3];
  const float* Wg     = (const float*)d_in[4];
  const float* Wr     = (const float*)d_in[5];
  const float* br     = (const float*)d_in[6];
  const float* Wa     = (const float*)d_in[7];
  const float* ba     = (const float*)d_in[8];
  const float* Wm     = (const float*)d_in[9];
  const float* bmlp   = (const float*)d_in[10];
  const float* g      = (const float*)d_in[11];
  const float* be     = (const float*)d_in[12];
  const float* Wrel   = (const float*)d_in[13];
  // d_in[14] = num_heads (16, hardcoded per setup)

  constexpr int Bb = 2048, U = 1024, S = 9;
  constexpr int BS = Bb * S;        // 18432 rows (mem_plus)
  constexpr int BM = Bb * 8;        // 16384 rows (memory)

  char* ws = (char*)d_ws;
  size_t off = 0;
  auto alloc = [&](size_t bytes) -> void* {
    void* p = ws + off;
    off += (bytes + 255) & ~(size_t)255;
    return p;
  };
  // bf16 operands (weights stored TRANSPOSED: N x K)
  __bf16* wi_bf  = (__bf16*)alloc((size_t)1024 * 1024 * 2);
  __bf16* wg_bf  = (__bf16*)alloc((size_t)1024 * 2048 * 2);
  __bf16* wr_bf  = (__bf16*)alloc((size_t)1024 * 2048 * 2);
  __bf16* wa_bf  = (__bf16*)alloc((size_t)1024 * 3072 * 2);
  __bf16* w1_bf  = (__bf16*)alloc((size_t)1024 * 1024 * 2);
  __bf16* w2_bf  = (__bf16*)alloc((size_t)1024 * 1024 * 2);
  __bf16* in_bf  = (__bf16*)alloc((size_t)Bb * U * 2);
  __bf16* mem_bf = (__bf16*)alloc((size_t)BM * U * 2);
  __bf16* mp_bf  = (__bf16*)alloc((size_t)BS * U * 2);
  __bf16* h_bf   = (__bf16*)alloc((size_t)BS * U * 2);
  __bf16* m1_bf  = (__bf16*)alloc((size_t)BS * U * 2);
  // f32 intermediates
  float* x_f   = (float*)alloc((size_t)Bb * U * 4);
  float* gx_f  = (float*)alloc((size_t)Bb * 2048 * 4);
  float* gr_f  = (float*)alloc((size_t)BM * 2048 * 4);
  float* mp_f  = (float*)alloc((size_t)BS * U * 4);
  float* qkv_f = (float*)alloc((size_t)BS * 3072 * 4);
  float* rel_f = (float*)alloc((size_t)17 * 1024 * 4);
  float* ao_f  = (float*)alloc((size_t)BS * U * 4);
  float* h_f   = (float*)alloc((size_t)BS * U * 4);
  float* m2_f  = (float*)alloc((size_t)BS * U * 4);
  float* h2_f  = (float*)alloc((size_t)BS * U * 4);

  // 1) casts (weights transposed) + rel table GEMM
  cvt_f32_bf16_T<<<256, 256, 0, stream>>>(Wi, wi_bf, 1024, 1024);
  cvt_f32_bf16_T<<<256, 256, 0, stream>>>(Wg, wg_bf, 1024, 2048);
  cvt_f32_bf16_T<<<256, 256, 0, stream>>>(Wr, wr_bf, 1024, 2048);
  cvt_f32_bf16_T<<<256, 256, 0, stream>>>(Wa, wa_bf, 1024, 3072);
  cvt_split_mlp_T<<<256, 256, 0, stream>>>(Wm, w1_bf, w2_bf);
  cvt_f32_bf16<<<256, 256, 0, stream>>>(inputs, in_bf, Bb * U);
  rel_emb_kernel<<<17, 256, 0, stream>>>(Wrel, rel_f);

  // 2) x = inputs@Wi + bi ; gates_x = inputs@Wg
  gemm_bf16_wmma<<<dim3(1024 / 64, Bb / 128), 128, 0, stream>>>(
      in_bf, wi_bf, bi, x_f, nullptr, Bb, 1024, 1024, 0);
  gemm_bf16_wmma<<<dim3(2048 / 64, Bb / 128), 128, 0, stream>>>(
      in_bf, wg_bf, nullptr, gx_f, nullptr, Bb, 2048, 1024, 0);

  // 3) mem_plus pack (+ bf16 memory copy)
  pack_memplus<<<4096, 256, 0, stream>>>(state, x_f, mp_f, mp_bf, mem_bf, BS * U);

  // 4) gates_rec = memory@Wr ; qkv = mem_plus@Wa + ba
  gemm_bf16_wmma<<<dim3(2048 / 64, BM / 128), 128, 0, stream>>>(
      mem_bf, wr_bf, nullptr, gr_f, nullptr, BM, 2048, 1024, 0);
  gemm_bf16_wmma<<<dim3(3072 / 64, BS / 128), 128, 0, stream>>>(
      mp_bf, wa_bf, ba, qkv_f, nullptr, BS, 3072, 1024, 0);

  // 5) attention
  attn_kernel<<<dim3(16, Bb), 128, 0, stream>>>(qkv_f, rel_f, ao_f);

  // 6) h = LN(mem_plus + att_out) with g1/b1 (f32 + bf16)
  ln_res_kernel<<<BS, 256, 0, stream>>>(mp_f, ao_f, g, be, h_f, h_bf, nullptr);

  // 7) MLP: gelu(h@w1 + mb1) @ w2 + mb2
  gemm_bf16_wmma<<<dim3(1024 / 64, BS / 128), 128, 0, stream>>>(
      h_bf, w1_bf, bmlp, nullptr, m1_bf, BS, 1024, 1024, 1);
  gemm_bf16_wmma<<<dim3(1024 / 64, BS / 128), 128, 0, stream>>>(
      m1_bf, w2_bf, bmlp + 1024, m2_f, nullptr, BS, 1024, 1024, 0);

  // 8) h2 = LN(h + mlp_out) with g2/b2; row s==8 goes straight to output
  float* out = (float*)d_out;
  ln_res_kernel<<<BS, 256, 0, stream>>>(h_f, m2_f, g + 1024, be + 1024, h2_f, nullptr, out);

  // 9) gates + next_state
  gate_fuse_kernel<<<2048, 256, 0, stream>>>(gx_f, gr_f, br, state, h2_f,
                                             out + (size_t)Bb * U, BM * U);
}
